// SpikingNeuralNetwork_53996328845632
// MI455X (gfx1250) — matrix operand order repaired and checked by reading, not compiled, not run on hardware
//
#include <hip/hip_runtime.h>

// ---------------------------------------------------------------------------
// Spiking NN: 100 recurrent steps of [4096,1024]x[1024,4096] + [4096,4096]x
// [4096,1024] GEMMs with LIF updates. Persistent tile-of-batch workgroups,
// bf16 WMMA (v_wmma_f32_16x16x32_bf16) with f32 accumulation, software-
// pipelined operand loads.
// ---------------------------------------------------------------------------

typedef __attribute__((ext_vector_type(16))) __bf16 v16bf;
typedef __attribute__((ext_vector_type(8)))  __bf16 v8bf;
typedef __attribute__((ext_vector_type(8)))  float  v8f;

#define B_TOT    4096
#define D_IN     1024
#define D_H      4096
#define D_OUT    1024
#define T_STEPS  100
#define BM       32                 // batch rows per workgroup
#define CH       256                // hidden-chunk width
#define NCH      (D_H / CH)         // 16 chunks per step
#define NTHREADS 256                // 8 wave32s

__device__ __forceinline__ v8f wmma_bf16(v16bf a, v16bf b, v8f c) {
  // (neg_a, A, neg_b, B, c_mod, C, reuse_a, reuse_b)
  return __builtin_amdgcn_wmma_f32_16x16x32_bf16(false, a, false, b,
                                                 (short)0, c, false, false);
}

// A-fragment (16x32 bf16 tile from a row-major source):
// lane m=lane&15, hi=lane>>4; elements 0..15 -> K=kb+hi*16+j
__device__ __forceinline__ v16bf load_a_frag(const __bf16* row_base, int k) {
  v8bf lo  = *(const v8bf*)(row_base + k);
  v8bf hi8 = *(const v8bf*)(row_base + k + 8);
  return __builtin_shufflevector(lo, hi8, 0,1,2,3,4,5,6,7,8,9,10,11,12,13,14,15);
}

__global__ void cvt_f32_bf16(const float* __restrict__ s, __bf16* __restrict__ d, int n) {
  int i = blockIdx.x * blockDim.x + threadIdx.x;
  if (i < n) d[i] = (__bf16)s[i];
}

__global__ __launch_bounds__(NTHREADS)
void snn_wmma_kernel(const __bf16* __restrict__ W1b,   // [D_H,  D_IN] bf16
                     const __bf16* __restrict__ W2b,   // [D_OUT, D_H] bf16
                     const float*  __restrict__ x,     // [B, D_IN] f32
                     const float*  __restrict__ b1,    // [D_H]
                     const float*  __restrict__ b2,    // [D_OUT]
                     float* __restrict__ memW,         // [B, D_OUT] membrane state (ws)
                     float* __restrict__ out)          // [B, D_OUT] spike accum / output
{
  __shared__ __align__(16) __bf16 sX[BM * D_IN];       // 64 KB: recurrent input, bf16
  __shared__ __align__(16) __bf16 sH[2][BM * CH];      // 2x16 KB: hidden chunk, dbl-buffered

  const int tid  = threadIdx.x;
  const int lane = tid & 31;
  const int wid  = tid >> 5;
  const int mv   = lane & 15;          // N-col / M-row inside a 16x16 tile
  const int hi   = lane >> 4;          // lane half
  const int row0 = blockIdx.x * BM;

  // GEMM2 region of this wave: 16 rows x 256 cols of the 32x1024 output tile
  const int rowHalf = wid >> 2;        // 0..1
  const int colQ    = wid & 3;         // 0..3

  // Stage x tile (f32 -> bf16) into LDS.
  for (int i = tid; i < BM * D_IN; i += NTHREADS) {
    sX[i] = (__bf16)x[(size_t)(row0 + (i >> 10)) * D_IN + (i & (D_IN - 1))];
  }
  __syncthreads();

  // Per-lane output-column biases (16 tiles of 16 cols).
  float b2r[16];
  #pragma unroll
  for (int t = 0; t < 16; ++t) b2r[t] = b2[colQ * 256 + t * 16 + mv];

  // Wave-invariant base pointers.
  const __bf16* x0 = sX + (0 * 16 + mv) * D_IN + hi * 16;
  const __bf16* x1 = sX + (1 * 16 + mv) * D_IN + hi * 16;
  const __bf16* w2base0 = W2b + (size_t)(colQ * 256 + mv) * D_H + hi * 16;

  for (int step = 0; step < T_STEPS; ++step) {
    // Y accumulators: 16 tiles x v8f = 128 VGPRs/lane, live across all chunks.
    v8f yacc[16];
    #pragma unroll
    for (int t = 0; t < 16; ++t) {
      v8f z = {0.f,0.f,0.f,0.f,0.f,0.f,0.f,0.f};
      yacc[t] = z;
    }

    for (int c = 0; c < NCH; ++c) {
      __bf16* hbuf = sH[c & 1];

      // ---------------- GEMM1: h = relu(x W1^T + b1), this chunk ------------
      // Wave computes col-tiles ct = wid*2 + {0,1}, both row-tiles each.
      #pragma unroll
      for (int ct2 = 0; ct2 < 2; ++ct2) {
        const int ct = wid * 2 + ct2;                 // 0..15 within chunk
        const int hb = c * CH + ct * 16;              // global hidden col base
        v8f acc0 = {0.f,0.f,0.f,0.f,0.f,0.f,0.f,0.f};
        v8f acc1 = {0.f,0.f,0.f,0.f,0.f,0.f,0.f,0.f};
        const __bf16* wrow = W1b + (size_t)(hb + mv) * D_IN + hi * 16;

        // software pipeline: fragments for k are loaded one iteration ahead
        v16bf bb = *(const v16bf*)(wrow);
        v16bf a0 = load_a_frag(x0, 0);
        v16bf a1 = load_a_frag(x1, 0);
        #pragma unroll 4
        for (int k = 0; k < D_IN - 32; k += 32) {
          v16bf bbn = *(const v16bf*)(wrow + k + 32);
          v16bf a0n = load_a_frag(x0, k + 32);
          v16bf a1n = load_a_frag(x1, k + 32);
          acc0 = wmma_bf16(a0, bb, acc0);
          acc1 = wmma_bf16(a1, bb, acc1);
          bb = bbn; a0 = a0n; a1 = a1n;
        }
        acc0 = wmma_bf16(a0, bb, acc0);               // peeled last k-step
        acc1 = wmma_bf16(a1, bb, acc1);

        const float bias = b1[hb + mv];
        const int   hc   = ct * 16 + mv;              // col within chunk
        #pragma unroll
        for (int r = 0; r < 8; ++r) {
          float h0 = acc0[r] + bias; h0 = h0 > 0.f ? h0 : 0.f;
          float h1 = acc1[r] + bias; h1 = h1 > 0.f ? h1 : 0.f;
          hbuf[( 0 + r + 8 * hi) * CH + hc] = (__bf16)h0;
          hbuf[(16 + r + 8 * hi) * CH + hc] = (__bf16)h1;
        }
      }
      __syncthreads();

      // ---------------- GEMM2 partial: y += h W2^T over this K-chunk --------
      const __bf16* hrow  = hbuf + (rowHalf * 16 + mv) * CH + hi * 16;
      const __bf16* wbase = w2base0 + c * CH;         // + t*16*D_H as imm offset
      v16bf a = load_a_frag(hrow, 0);
      #pragma unroll
      for (int k = 0; k < CH; k += 32) {
        v16bf an;
        if (k + 32 < CH) an = load_a_frag(hrow, k + 32);   // hoist next-A early
        #pragma unroll
        for (int t = 0; t < 16; ++t) {
          v16bf bb = *(const v16bf*)(wbase + k + (size_t)t * (16 * D_H));
          yacc[t] = wmma_bf16(a, bb, yacc[t]);
        }
        a = an;
      }
      // double-buffering of sH + per-chunk barrier covers WAR across waves
    }

    // ---------------- LIF neuron update + recurrent feedback ---------------
    const float inv = 1.0f / (float)T_STEPS;
    #pragma unroll
    for (int t = 0; t < 16; ++t) {
      const int ob = colQ * 256 + t * 16;
      #pragma unroll
      for (int r = 0; r < 8; ++r) {
        const int lr = rowHalf * 16 + r + 8 * hi;                 // local row 0..31
        const size_t gi = (size_t)(row0 + lr) * D_OUT + ob + mv;
        float y     = yacc[t][r] + b2r[t];
        float mOld  = (step == 0) ? 0.0f : memW[gi];
        float spOld = (step == 0) ? 0.0f : out[gi];
        float m2 = mOld * 0.2f + y;                                // DECAY = 0.2
        float s  = (m2 >= 0.5f) ? 1.0f : 0.0f;                     // THRESHOLD = 0.5
        float sp = spOld + s;
        memW[gi] = m2 * (1.0f - s);
        out[gi]  = (step == T_STEPS - 1) ? sp * inv : sp;
        sX[lr * D_IN + ob + mv] = (__bf16)y;                       // y_t -> x_{t+1}
      }
    }
    __syncthreads();
  }
}

extern "C" void kernel_launch(void* const* d_in, const int* in_sizes, int n_in,
                              void* d_out, int out_size, void* d_ws, size_t ws_size,
                              hipStream_t stream)
{
  const float* x  = (const float*)d_in[0];
  const float* W1 = (const float*)d_in[1];
  const float* b1 = (const float*)d_in[2];
  const float* W2 = (const float*)d_in[3];
  const float* b2 = (const float*)d_in[4];
  // d_in[5] = time_steps (==100 per reference; hardcoded)

  __bf16* W1b  = (__bf16*)d_ws;                          //  8 MB
  __bf16* W2b  = W1b + (size_t)D_H * D_IN;               //  8 MB
  float*  memW = (float*)(W2b + (size_t)D_OUT * D_H);    // 16 MB membrane state

  const int n1 = D_H * D_IN, n2 = D_OUT * D_H;
  cvt_f32_bf16<<<(n1 + 255) / 256, 256, 0, stream>>>(W1, W1b, n1);
  cvt_f32_bf16<<<(n2 + 255) / 256, 256, 0, stream>>>(W2, W2b, n2);

  snn_wmma_kernel<<<B_TOT / BM, NTHREADS, 0, stream>>>(
      W1b, W2b, x, b1, b2, memW, (float*)d_out);
}
